// FullAttention_v43_47330539601997
// MI455X (gfx1250) — matrix operand order repaired and checked by reading, not compiled, not run on hardware
//
#include <hip/hip_runtime.h>

typedef __attribute__((ext_vector_type(16))) _Float16 v16h;
typedef __attribute__((ext_vector_type(8)))  _Float16 v8h;
typedef __attribute__((ext_vector_type(8)))  float    v8f;

// Problem constants (from reference): B=4, L=S=2048, H=8, E=D=64
constexpr int Bc = 4, Lc = 2048, Sc = 2048, Hc = 8, Ec = 64, Dc = 64;
constexpr int LDSS = 72;   // padded LDS row stride in halves (144B, 16B-aligned)

// scale folded into Q: (1/sqrt(E)) * log2(e) -> softmax done in exp2 domain
#define QSCALE 0.18033688011112042f   // 0.125 * 1.4426950408889634

// Build a v16h WMMA operand from two 16-byte LDS reads.
__device__ __forceinline__ v16h make_op(const _Float16* lo_p, const _Float16* hi_p) {
  v8h lo = *(const v8h*)lo_p;
  v8h hi = *(const v8h*)hi_p;
  v16h r;
#pragma unroll
  for (int i = 0; i < 8; ++i) { r[i] = lo[i]; r[i + 8] = hi[i]; }
  return r;
}

// Max-reduce across each 16-lane DPP row (lanes 0-15 / 16-31 independently)
// using DPP-fused VOP2 max: 4 VALU instructions total.
__device__ __forceinline__ float rowmax16_dpp(float x) {
  float r;
  asm("v_max_num_f32_dpp %0, %1, %1 quad_perm:[1,0,3,2] row_mask:0xf bank_mask:0xf bound_ctrl:1"
      : "=v"(r) : "v"(x));
  asm("v_max_num_f32_dpp %0, %1, %1 quad_perm:[2,3,0,1] row_mask:0xf bank_mask:0xf bound_ctrl:1"
      : "=v"(x) : "v"(r));
  asm("v_max_num_f32_dpp %0, %1, %1 row_half_mirror row_mask:0xf bank_mask:0xf bound_ctrl:1"
      : "=v"(r) : "v"(x));
  asm("v_max_num_f32_dpp %0, %1, %1 row_mirror row_mask:0xf bank_mask:0xf bound_ctrl:1"
      : "=v"(x) : "v"(r));
  return x;
}

__global__ __launch_bounds__(256)
void fa_wmma_kernel(const float* __restrict__ Q, const float* __restrict__ K,
                    const float* __restrict__ V, const float* __restrict__ M,
                    float* __restrict__ O) {
  __shared__ __align__(16) _Float16 Klds[64 * LDSS];        // K tile: [s][e]
  __shared__ __align__(16) _Float16 Vt[64 * LDSS];          // V tile transposed: [d][s]
  __shared__ __align__(16) _Float16 P[8 * 16 * LDSS];       // per-wave probs: [m][s]

  const int lt   = blockIdx.x;           // L tile (128 rows)
  const int h    = blockIdx.y;
  const int b    = blockIdx.z;
  const int wave = threadIdx.x >> 5;
  const int lane = threadIdx.x & 31;
  const int hf   = lane >> 4;            // lane half (WMMA K/M split)
  const int ln   = lane & 15;
  const int l0   = lt * 128;

  // ---- Q A-operands in registers, pre-scaled by 0.125*log2e (fp32->f16) ---
  // A layout: lane row = ln; element t of 32-chunk: e = (t/8)*16 + hf*8 + t%8
  const int qrow = l0 + wave * 16 + ln;
  const float* qp = Q + (((size_t)b * Lc + qrow) * Hc + h) * Ec;
  v16h a0, a1;
#pragma unroll
  for (int t = 0; t < 16; ++t) {
    const int e = ((t >> 3) << 4) + (hf << 3) + (t & 7);
    a0[t] = (_Float16)(qp[e]      * QSCALE);
    a1[t] = (_Float16)(qp[32 + e] * QSCALE);
  }

  // all-ones B operand for row-sum-of-P WMMA
  v16h ones;
#pragma unroll
  for (int t = 0; t < 16; ++t) ones[t] = (_Float16)1.0f;

  // ---- online-softmax state (log2 domain) ---------------------------------
  float mrow[8], lrow[8];
  v8f acc[4];
#pragma unroll
  for (int i = 0; i < 8; ++i) { mrow[i] = -1e30f; lrow[i] = 0.f; }
#pragma unroll
  for (int g = 0; g < 4; ++g)
#pragma unroll
    for (int i = 0; i < 8; ++i) acc[g][i] = 0.f;

  // cooperative-load mapping: 256 thr -> 64 rows x 4 groups of 16 cols
  const int ldr = threadIdx.x >> 2;
  const int ldc = (threadIdx.x & 3) << 4;
  const float* kp = K + (((size_t)b * Sc + ldr) * Hc + h) * Ec + ldc;
  const float* vp = V + (((size_t)b * Sc + ldr) * Hc + h) * Dc + ldc;
  const size_t srow = (size_t)Hc * Ec;   // 512 floats per s-step

  _Float16* pw = &P[wave * 16 * LDSS];   // this wave's P region

  for (int s0 = 0; s0 < Sc; s0 += 64) {
    __syncthreads();                     // previous tile fully consumed
    {
      const float* kk = kp + (size_t)s0 * srow;
      v8h klo, khi;
#pragma unroll
      for (int j = 0; j < 8; ++j) { klo[j] = (_Float16)kk[j]; khi[j] = (_Float16)kk[8 + j]; }
      *(v8h*)&Klds[ldr * LDSS + ldc]     = klo;
      *(v8h*)&Klds[ldr * LDSS + ldc + 8] = khi;
      const float* vv = vp + (size_t)s0 * srow;
#pragma unroll
      for (int j = 0; j < 16; ++j)       // transpose V on the way in
        Vt[(ldc + j) * LDSS + ldr] = (_Float16)vv[j];
    }
    if (s0 + 64 < Sc) {                  // CDNA5 global_prefetch_b8 for next tile
      __builtin_prefetch(kp + (size_t)(s0 + 64) * srow, 0, 3);
      __builtin_prefetch(vp + (size_t)(s0 + 64) * srow, 0, 3);
    }
    __syncthreads();

    // ---- preload all K B-operands (8 ds_load_b128 pairs, distinct regs) ---
    v16h kb0[4], kb1[4];
#pragma unroll
    for (int g = 0; g < 4; ++g) {
      const _Float16* kr = &Klds[(g * 16 + ln) * LDSS + hf * 8];
      kb0[g] = make_op(kr,      kr + 16);    // e = 0..31 slice
      kb1[g] = make_op(kr + 32, kr + 48);    // e = 32..63 slice
    }

    // ---- scores: S = Q * K^T, then multiplicative mask (log2 domain) ------
    v8f x[4];
    const int mrbase = l0 + wave * 16 + hf * 8;
#pragma unroll
    for (int g = 0; g < 4; ++g) {
      v8f c;
#pragma unroll
      for (int i = 0; i < 8; ++i) c[i] = 0.f;
      c = __builtin_amdgcn_wmma_f32_16x16x32_f16(false, a0, false, kb0[g], (short)0, c, false, false);
      c = __builtin_amdgcn_wmma_f32_16x16x32_f16(false, a1, false, kb1[g], (short)0, c, false, false);
#pragma unroll
      for (int i = 0; i < 8; ++i) {
        const float mv = M[(size_t)(mrbase + i) * Sc + (s0 + g * 16 + ln)];
        x[g][i] = mv * c[i];               // scale+log2e already folded into Q
      }
    }

    // ---- online softmax: row max via fused DPP butterfly ------------------
    float alpha[8];
#pragma unroll
    for (int i = 0; i < 8; ++i) {
      float vmax = fmaxf(fmaxf(x[0][i], x[1][i]), fmaxf(x[2][i], x[3][i]));
      vmax = rowmax16_dpp(vmax);
      const float mnew = fmaxf(mrow[i], vmax);
      alpha[i] = __builtin_amdgcn_exp2f(mrow[i] - mnew);
      mrow[i]  = mnew;
#pragma unroll
      for (int g = 0; g < 4; ++g) x[g][i] = __builtin_amdgcn_exp2f(x[g][i] - mnew);
    }
#pragma unroll
    for (int g = 0; g < 4; ++g)
#pragma unroll
      for (int i = 0; i < 8; ++i) acc[g][i] *= alpha[i];

    // ---- stage P (f16) through wave-private LDS, reload in A layout -------
#pragma unroll
    for (int i = 0; i < 8; ++i)
#pragma unroll
      for (int g = 0; g < 4; ++g)
        pw[(hf * 8 + i) * LDSS + g * 16 + ln] = (_Float16)x[g][i];
    asm volatile("s_wait_dscnt 0" ::: "memory");   // wave-local store->load fence

    const _Float16* pr = &pw[ln * LDSS];
    v16h p0 = make_op(pr + hf * 8,       pr + 16 + hf * 8);  // s = 0..31
    v16h p1 = make_op(pr + 32 + hf * 8,  pr + 48 + hf * 8);  // s = 32..63

    // ---- preload all V B-operands (loads overlap row-sum WMMA below) ------
    v16h vb0[4], vb1[4];
#pragma unroll
    for (int g = 0; g < 4; ++g) {
      const _Float16* vr = &Vt[(g * 16 + ln) * LDSS + hf * 8];
      vb0[g] = make_op(vr,      vr + 16);
      vb1[g] = make_op(vr + 32, vr + 48);
    }

    // ---- row sums of P via ones-matrix WMMA -------------------------------
    // result element i = sum_s P[hf*8+i][s], identical in every lane
    {
      v8f rs;
#pragma unroll
      for (int i = 0; i < 8; ++i) rs[i] = 0.f;
      rs = __builtin_amdgcn_wmma_f32_16x16x32_f16(false, p0, false, ones, (short)0, rs, false, false);
      rs = __builtin_amdgcn_wmma_f32_16x16x32_f16(false, p1, false, ones, (short)0, rs, false, false);
#pragma unroll
      for (int i = 0; i < 8; ++i) lrow[i] = lrow[i] * alpha[i] + rs[i];
    }

    // ---- O += P * V (2 chained WMMAs per 16-col D group) ------------------
#pragma unroll
    for (int g = 0; g < 4; ++g) {
      acc[g] = __builtin_amdgcn_wmma_f32_16x16x32_f16(false, p0, false, vb0[g], (short)0, acc[g], false, false);
      acc[g] = __builtin_amdgcn_wmma_f32_16x16x32_f16(false, p1, false, vb1[g], (short)0, acc[g], false, false);
    }
  }

  // ---- epilogue: divide by row sums, store fp32 ---------------------------
#pragma unroll
  for (int i = 0; i < 8; ++i) {
    const float inv  = 1.0f / lrow[i];
    const int   orow = l0 + wave * 16 + hf * 8 + i;
    float* op = O + (((size_t)b * Lc + orow) * Hc + h) * Dc;
#pragma unroll
    for (int g = 0; g < 4; ++g) op[g * 16 + ln] = acc[g][i] * inv;
  }
}

extern "C" void kernel_launch(void* const* d_in, const int* in_sizes, int n_in,
                              void* d_out, int out_size, void* d_ws, size_t ws_size,
                              hipStream_t stream) {
  (void)in_sizes; (void)n_in; (void)d_ws; (void)ws_size; (void)out_size;
  const float* q = (const float*)d_in[0];
  const float* k = (const float*)d_in[1];
  const float* v = (const float*)d_in[2];
  const float* m = (const float*)d_in[3];
  float* out = (float*)d_out;
  dim3 grid(Lc / 128, Hc, Bc);   // (16, 8, 4) blocks of 256 threads (8 wave32)
  fa_wmma_kernel<<<grid, 256, 0, stream>>>(q, k, v, m, out);
}
